// BaseModel_87780541595879
// MI455X (gfx1250) — matrix operand order repaired and checked by reading, not compiled, not run on hardware
//
#include <hip/hip_runtime.h>
#include <math.h>

typedef _Float16 v16h __attribute__((ext_vector_type(16)));
typedef float    v8f  __attribute__((ext_vector_type(8)));

#define TN   32
#define PN   16
#define NPTS 65536

// ---------------------------------------------------------------------------
// Kernel A: 6D -> rotation matrices (T*P = 512 of them).
// Writes [R(9), t(3)] per (t,p) into workspace (stride 12 floats) for kernel B,
// and the (T,P,4,4) trans_list directly into d_out.
// ---------------------------------------------------------------------------
__global__ void rot6d_kernel(const float* __restrict__ d6,
                             const float* __restrict__ tvec,
                             float* __restrict__ Rws,
                             float* __restrict__ trans_out)
{
    int id = blockIdx.x * blockDim.x + threadIdx.x;   // 0..511 = t*16+p
    if (id >= TN * PN) return;
    const float* a = d6 + id * 6;
    float a1x = a[0], a1y = a[1], a1z = a[2];
    float a2x = a[3], a2y = a[4], a2z = a[5];
    float n1 = sqrtf(a1x*a1x + a1y*a1y + a1z*a1z);
    float b1x = a1x/n1, b1y = a1y/n1, b1z = a1z/n1;
    float d  = b1x*a2x + b1y*a2y + b1z*a2z;
    float px = a2x - d*b1x, py = a2y - d*b1y, pz = a2z - d*b1z;
    float n2 = sqrtf(px*px + py*py + pz*pz);
    float b2x = px/n2, b2y = py/n2, b2z = pz/n2;
    float b3x = b1y*b2z - b1z*b2y;
    float b3y = b1z*b2x - b1x*b2z;
    float b3z = b1x*b2y - b1y*b2x;
    float R[9] = { b1x,b1y,b1z,  b2x,b2y,b2z,  b3x,b3y,b3z };

    const float* tv = tvec + id * 3;
    float* w = Rws + id * 12;
    #pragma unroll
    for (int j = 0; j < 9; j++) w[j] = R[j];
    w[9] = tv[0]; w[10] = tv[1]; w[11] = tv[2];

    float* to = trans_out + id * 16;           // (T,P,4,4)
    #pragma unroll
    for (int j = 0; j < 3; j++) {
        to[j*4+0] = R[j*3+0]; to[j*4+1] = R[j*3+1];
        to[j*4+2] = R[j*3+2]; to[j*4+3] = tv[j];
    }
    to[12] = 0.f; to[13] = 0.f; to[14] = 0.f; to[15] = 1.f;
}

// ---------------------------------------------------------------------------
// Kernel B: fused  h=relu(pc@w1T) -> seg=h@w2T (WMMA) -> dual argmax ->
//           per-point rigid transform for all 32 frames -> pc_out stream.
// 256 threads = 8 waves; each wave owns a 16-point tile.
// ---------------------------------------------------------------------------
__global__ void __launch_bounds__(256)
fused_kernel(const float* __restrict__ pc,
             const float* __restrict__ w1,
             const float* __restrict__ b1,
             const float* __restrict__ w2,
             const float* __restrict__ b2,
             const float* __restrict__ gumbel,
             const float* __restrict__ Rws,
             float* __restrict__ pc_out,
             float* __restrict__ amax_out)
{
    __shared__ float Rt[TN * PN * 12];      // 24 KB: rotation+t table, block-local
    __shared__ float segs[8][16][16];       // 8 KB:  per-wave seg tile exchange
    __shared__ int   psel_s[8][16];         // selected part per point

    const int tid  = threadIdx.x;
    const int wave = tid >> 5;
    const int lane = tid & 31;
    const int m    = lane & 15;             // A-frag row (point) / B & C column
    const int hi   = lane >> 4;             // half-wave selector in WMMA layouts

    // Stage rotation table into LDS (coalesced, 24 floats/thread).
    #pragma unroll
    for (int i = 0; i < 24; i++) Rt[tid + 256 * i] = Rws[tid + 256 * i];

    const int base = blockIdx.x * 128 + wave * 16;  // 65536 % 128 == 0: no tail
    const int n    = base + m;
    const float x0 = pc[n*3+0], x1 = pc[n*3+1], x2 = pc[n*3+2];

    // C init in the 16x16 f32 C/D layout: acc[r] = seg[M=r+8*hi][N=lane&15],
    // so every slot starts at b2[column].
    v8f acc;
    {
        float bb = b2[m];
        #pragma unroll
        for (int r = 0; r < 8; r++) acc[r] = bb;
    }

    // seg(16x16) = h(16x128) @ w2T(128x16): 4 chained v_wmma_f32_16x16x32_f16.
    #pragma unroll
    for (int c = 0; c < 4; c++) {
        v16h af, bf;
        // A frag (16x32 f16): lane holds row m; K index per the ISA table:
        //   k = (v<4 ? 2v : 2v+8) + 8*hi + (e&1), v = e/2, plus 32*c chunk base.
        #pragma unroll
        for (int e = 0; e < 16; e++) {
            int v = e >> 1;
            int k = 32*c + ((v < 4) ? (2*v) : (2*v + 8)) + 8*hi + (e & 1);
            float hv = b1[k] + x0*w1[k*3+0] + x1*w1[k*3+1] + x2*w1[k*3+2];
            hv = fmaxf(hv, 0.0f);                       // relu, computed in-place
            af[e] = (_Float16)hv;
        }
        // B frag (32x16 f16): lane holds column m; K order matches A:
        //   k = (e<8 ? e : e+8) + 8*hi within the chunk. B[k][p] = w2[p][k].
        #pragma unroll
        for (int e = 0; e < 16; e++) {
            int k = 32*c + ((e < 8) ? e : (e + 8)) + 8*hi;
            bf[e] = (_Float16)w2[m*128 + k];
        }
        acc = __builtin_amdgcn_wmma_f32_16x16x32_f16(
                  false, af, false, bf, (short)0, acc, false, false);
    }

    // Scatter seg tile to LDS: acc[r] -> point (r+8*hi), part (lane&15).
    #pragma unroll
    for (int r = 0; r < 8; r++) segs[wave][r + 8*hi][m] = acc[r];

    __syncthreads();

    // Dual argmax over the 16 parts (first-max semantics like jnp.argmax).
    // weight == hard one-hot in the forward pass, so softmax is never needed.
    if (hi == 0) {
        float bs = -3.4e38f, bsg = -3.4e38f;
        int pa = 0, pg = 0;
        #pragma unroll
        for (int p = 0; p < 16; p++) {
            float s  = segs[wave][m][p];
            float sg = s + gumbel[n * 16 + p];
            if (s  > bs ) { bs  = s;  pa = p; }
            if (sg > bsg) { bsg = sg; pg = p; }
        }
        psel_s[wave][m] = pg;
        amax_out[n] = (float)pa;                 // output 2: argmax(seg)
    }
    __syncthreads();

    // Apply the selected rigid transform for 16 frames per half-wave.
    const int ps = psel_s[wave][m];
    #pragma unroll
    for (int tt = 0; tt < 16; tt++) {
        int t = tt + 16 * hi;
        const float* rw = &Rt[(t * PN + ps) * 12];
        float y0 = rw[0]*x0 + rw[1]*x1 + rw[2]*x2 + rw[9];
        float y1 = rw[3]*x0 + rw[4]*x1 + rw[5]*x2 + rw[10];
        float y2 = rw[6]*x0 + rw[7]*x1 + rw[8]*x2 + rw[11];
        float* o = pc_out + (size_t)(t * NPTS + n) * 3;
        // 24 MB streaming output: non-temporal so it doesn't thrash caches.
        __builtin_nontemporal_store(y0, o + 0);
        __builtin_nontemporal_store(y1, o + 1);
        __builtin_nontemporal_store(y2, o + 2);
    }
}

// ---------------------------------------------------------------------------
extern "C" void kernel_launch(void* const* d_in, const int* in_sizes, int n_in,
                              void* d_out, int out_size, void* d_ws, size_t ws_size,
                              hipStream_t stream)
{
    (void)in_sizes; (void)n_in; (void)out_size; (void)ws_size;
    const float* cano = (const float*)d_in[0];
    const float* w1   = (const float*)d_in[1];
    const float* b1   = (const float*)d_in[2];
    const float* w2   = (const float*)d_in[3];
    const float* b2   = (const float*)d_in[4];
    const float* p6d  = (const float*)d_in[5];
    const float* pt   = (const float*)d_in[6];
    const float* gum  = (const float*)d_in[7];

    float* out    = (float*)d_out;
    float* pc_out = out;                                 // (T, N, 3)
    float* amax   = out + (size_t)TN * NPTS * 3;         // (N,)
    float* trans  = amax + NPTS;                         // (T, P, 4, 4)
    float* Rws    = (float*)d_ws;                        // 512 * 12 floats

    rot6d_kernel<<<2, 256, 0, stream>>>(p6d, pt, Rws, trans);
    fused_kernel<<<NPTS / 128, 256, 0, stream>>>(cano, w1, b1, w2, b2, gum,
                                                 Rws, pc_out, amax);
}
